// SinkhornUnmatched_58033598103848
// MI455X (gfx1250) — matrix operand order repaired and checked by reading, not compiled, not run on hardware
//
#include <hip/hip_runtime.h>
#include <hip/hip_bf16.h>

// CDNA5 / gfx1250 wave32 WMMA types
typedef __attribute__((ext_vector_type(16))) _Float16 v16h;
typedef __attribute__((ext_vector_type(8)))  float    v8f;

#define SK_EPS 1e-12f

// Wave-local LDS fence: LDS ops from one wave complete in order (ISA ch.11),
// so compiler ordering (wave_barrier) + draining DScnt handles intra-wave
// store->load / load->overwrite hazards. Cheaper than s_barrier.
__device__ __forceinline__ void wave_lds_fence() {
  __builtin_amdgcn_wave_barrier();
  asm volatile("s_wait_dscnt 0x0" ::: "memory");
  __builtin_amdgcn_wave_barrier();
}

// ---------------------------------------------------------------------------
// Setup kernel: pack the 4 weight matrices into 10 WMMA B-fragments (f16,
// 32x16 KxN, per-lane layout: lane<16 -> N=lane, K=k0+{0..7,16..23} in halves 0..15;
// lane>=16 -> N=lane-16, K=k0+{8..15,24..31}). Zero-padding for K/N overhang is
// baked in here ONCE, so the hot kernel has no lane-dependent bounds checks.
//   frag 0-3 : W1 (6x64),  k0=0,  n0=16*f
//   frag 4-5 : W2 (64x32), k0=0,  n0=16*(f-4)
//   frag 6-7 : W2 (64x32), k0=32, n0=16*(f-6)
//   frag 8   : W3 (32x16)
//   frag 9   : W4 (16x9)
// ---------------------------------------------------------------------------
__global__ void prep_frags_kernel(const float* __restrict__ W1,
                                  const float* __restrict__ W2,
                                  const float* __restrict__ W3,
                                  const float* __restrict__ W4,
                                  _Float16* __restrict__ wsB) {
  const int f = blockIdx.x;      // 0..9
  const int lane = threadIdx.x;  // 0..31
  const float* W; int fi, fo, k0, n0;
  if (f < 4)       { W = W1; fi = 6;  fo = 64; k0 = 0;  n0 = f * 16; }
  else if (f < 6)  { W = W2; fi = 64; fo = 32; k0 = 0;  n0 = (f - 4) * 16; }
  else if (f < 8)  { W = W2; fi = 64; fo = 32; k0 = 32; n0 = (f - 6) * 16; }
  else if (f == 8) { W = W3; fi = 32; fo = 16; k0 = 0;  n0 = 0; }
  else             { W = W4; fi = 16; fo = 9;  k0 = 0;  n0 = 0; }

  const int n   = n0 + (lane & 15);
  const int khi = (lane & 16) ? 8 : 0;
  v16h b;
#pragma unroll
  for (int e = 0; e < 16; ++e) {
    const int k = k0 + khi + (e & 7) + ((e >> 3) << 4);
    b[e] = (n < fo && k < fi) ? (_Float16)W[k * fo + n] : (_Float16)0.0f;
  }
  ((v16h*)wsB)[f * 32 + lane] = b;
}

// ---------------------------------------------------------------------------
// Scatter one 16x16 C tile (bias+ReLU, f16) into LDS laid out as the NEXT
// layer's A-fragments, so the consumer reads 2 contiguous ds_load_b128.
// Inverse of ISA 7.12.2 16-bit A layout:
//   element (M, k):  frag  = k>>5
//                    lane  = (M&15) + 16*((k>>3)&1)
//                    pos   = (k&7) + 8*((k>>4)&1)
// k = ncBase + (lane&15) is constant across the 8 accumulator VGPRs, so the
// 8 stores are a fixed 32B stride — pure address arithmetic, no branches.
// ---------------------------------------------------------------------------
__device__ __forceinline__ void scatter_relu_tile(_Float16* fragBase, const v8f c,
                                                  float bias, int k, int mb) {
  const int kf   = k >> 5;
  const int k5   = k & 31;
  const int lcHi = ((k5 >> 3) & 1) << 4;
  const int p    = (k5 & 7) | (((k5 >> 4) & 1) << 3);
  _Float16* dst = fragBase + (kf * 32 + lcHi) * 16 + p;
#pragma unroll
  for (int r = 0; r < 8; ++r)
    dst[(mb + r) * 16] = (_Float16)fmaxf(c[r] + bias, 0.0f);
}

__global__ __launch_bounds__(256) void sinkhorn_mlp_kernel(
    const float* __restrict__ margins,
    const float* __restrict__ b1, const float* __restrict__ b2,
    const float* __restrict__ b3, const float* __restrict__ b4,
    const _Float16* __restrict__ wsB,
    float* __restrict__ out, int B) {
  __shared__ float sM[256 * 6];                       // margins staging
  __shared__ float sB1[64], sB2[32], sB3[16], sB4[16];
  __shared__ __align__(32) _Float16 sFrag[8][2 * 32 * 16];  // per-wave A-frag staging
  __shared__ float sP[8][32 * 9];                     // per-wave MLP outputs

  const int tid  = threadIdx.x;
  const int lane = tid & 31;
  const int w    = tid >> 5;
  const long long blockBase = (long long)blockIdx.x * 256;

  // ---- cooperative staging (coalesced b32 loads) ----
  const long long nMarg = (long long)B * 6;
#pragma unroll 2
  for (int i = tid; i < 256 * 6; i += 256) {
    const long long g = blockBase * 6 + i;
    sM[i] = (g < nMarg) ? margins[g] : 0.0f;
  }
  if (tid < 64) sB1[tid] = b1[tid];
  if (tid < 32) sB2[tid] = b2[tid];
  if (tid < 16) sB3[tid] = b3[tid];
  if (tid < 16) sB4[tid] = (tid < 9) ? b4[tid] : 0.0f;
  __syncthreads();

  const int n_ = lane & 15;              // C/D layout: N = lane & 15
  const int mb = (lane & 16) ? 8 : 0;    //            M = mb + r
  _Float16* FB = &sFrag[w][0];
  const v16h* wsv = (const v16h*)wsB;    // 2x global_load_b128 per fragment
  const v8f z = {};

#pragma unroll
  for (int t = 0; t < 2; ++t) {
    const int rowbase = w * 32 + t * 16;

    // ---- Layer 1 A-fragment, branch-free ----
    // For K=6: halves 0..5 of lanes 0..15 hold x[m][0..5]; everything else 0.
    // Lanes 16..31 read too (uniform flow) but are masked to zero (v_cndmask).
    const float msk = (lane & 16) ? 0.0f : 1.0f;
    const float* xr = &sM[(rowbase + n_) * 6];
    v16h a1;
#pragma unroll
    for (int j = 0; j < 16; ++j) a1[j] = (_Float16)0.0f;
#pragma unroll
    for (int j = 0; j < 6; ++j) a1[j] = (_Float16)(xr[j] * msk);

    // ---- Layer 1: x(16x6) @ W1(6x64) -> scatter as layer-2 A-frags (K=64) ----
#pragma unroll
    for (int nc = 0; nc < 4; ++nc) {
      const v16h bf = wsv[nc * 32 + lane];
      const v8f c = __builtin_amdgcn_wmma_f32_16x16x32_f16(false, a1, false, bf,
                                                           (short)0, z, false, false);
      scatter_relu_tile(FB, c, sB1[nc * 16 + n_], nc * 16 + n_, mb);
    }
    wave_lds_fence();

    // ---- Layer 2: h1(16x64) @ W2(64x32), 2 K-steps x 2 N-chunks ----
    const v16h a2k0 = *(const v16h*)(FB + lane * 16);          // ds_load_b128 x2
    const v16h a2k1 = *(const v16h*)(FB + (32 + lane) * 16);
    v8f c2[2];
#pragma unroll
    for (int nc = 0; nc < 2; ++nc) {
      const v16h bk0 = wsv[(4 + nc) * 32 + lane];
      const v16h bk1 = wsv[(6 + nc) * 32 + lane];
      v8f c = __builtin_amdgcn_wmma_f32_16x16x32_f16(false, a2k0, false, bk0,
                                                     (short)0, z, false, false);
      c = __builtin_amdgcn_wmma_f32_16x16x32_f16(false, a2k1, false, bk1,
                                                 (short)0, c, false, false);
      c2[nc] = c;
    }
    wave_lds_fence();  // a2 loads drained before FB is overwritten
#pragma unroll
    for (int nc = 0; nc < 2; ++nc)  // h2 (K=32 exact) as layer-3 A-frag
      scatter_relu_tile(FB, c2[nc], sB2[nc * 16 + n_], nc * 16 + n_, mb);
    wave_lds_fence();

    // ---- Layer 3: h2(16x32) @ W3(32x16) ----
    const v16h a3 = *(const v16h*)(FB + lane * 16);
    const v16h b3f = wsv[8 * 32 + lane];
    const v8f c3 = __builtin_amdgcn_wmma_f32_16x16x32_f16(false, a3, false, b3f,
                                                          (short)0, z, false, false);
    wave_lds_fence();
    // h3 (16x16, K=16) as layer-4 A-frag in region 1: positions 8..15 must be
    // zero (K padding) -> one b128 zero-store per lane, then 8 b16 scatters.
    _Float16* FB1 = FB + 32 * 16;
    *(float4*)(FB1 + lane * 16 + 8) = make_float4(0.f, 0.f, 0.f, 0.f);
    {
      const float bias = sB3[n_];
      const int lcHi = ((n_ >> 3) & 1) << 4;   // k = n_ in 0..15, pos = k&7
      _Float16* dst = FB1 + lcHi * 16 + (n_ & 7);
#pragma unroll
      for (int r = 0; r < 8; ++r)
        dst[(mb + r) * 16] = (_Float16)fmaxf(c3[r] + bias, 0.0f);
    }
    wave_lds_fence();

    // ---- Layer 4: h3(16x16) @ W4(16x9) ----
    const v16h a4 = *(const v16h*)(FB1 + lane * 16);
    const v16h b4f = wsv[9 * 32 + lane];
    const v8f c4 = __builtin_amdgcn_wmma_f32_16x16x32_f16(false, a4, false, b4f,
                                                          (short)0, z, false, false);
    if (n_ < 9) {
      const float bias = sB4[n_];
#pragma unroll
      for (int r = 0; r < 8; ++r)
        sP[w][(t * 16 + mb + r) * 9 + n_] = c4[r] + bias;
    }
    wave_lds_fence();  // pars visible + FB free for next tile
  }

  // ---- per-lane epilogue: one row per lane (tau/exp/sigmoid + Sinkhorn x10) ----
  const int lrow = w * 32 + lane;
  const long long grow = blockBase + lrow;

  float p[9];
#pragma unroll
  for (int j = 0; j < 9; ++j) p[j] = sP[w][lane * 9 + j];
  const float M0 = sM[lrow * 6 + 0], M1 = sM[lrow * 6 + 1], M2 = sM[lrow * 6 + 2];
  const float F0 = sM[lrow * 6 + 3], F1 = sM[lrow * 6 + 4], F2 = sM[lrow * 6 + 5];

  // tau: muc = [[e^p0, e^p1, 1],[e^p2, e^p3, 1],[1,1,1]]
  float A00 = __expf(p[0]), A01 = __expf(p[1]);
  float A10 = __expf(p[2]), A11 = __expf(p[3]);
  float A02 = 1.0f, A12 = 1.0f, A20 = 1.0f, A21 = 1.0f, A22 = 1.0f;

  // sqs(x) = 0.02 + 0.96 * sigmoid(x)
  const float shm0 = 0.02f + 0.96f / (1.0f + __expf(-p[4]));
  const float shm1 = 0.02f + 0.96f / (1.0f + __expf(-p[5]));
  const float shf0 = 0.02f + 0.96f / (1.0f + __expf(-p[6]));
  const float shf1 = 0.02f + 0.96f / (1.0f + __expf(-p[7]));
  const float V = __expf(p[8]);

  const float r0 = M0 * shm0, r1 = M1 * shm1, r2 = M2;   // row targets
  const float c0 = F0 * shf0, c1 = F1 * shf1, c2 = F2;   // col targets
  const float mm0 = M0 - r0, mm1 = M1 - r1;              // mum0 (3rd comp = 0)
  const float ff0 = F0 - c0, ff1 = F1 - c1;              // mu0f (3rd comp = 0)

#pragma unroll
  for (int it = 0; it < 10; ++it) {
    const float s0 = r0 / (A00 + A01 + A02 + SK_EPS);
    const float s1 = r1 / (A10 + A11 + A12 + SK_EPS);
    const float s2 = r2 / (A20 + A21 + A22 + SK_EPS);
    A00 *= s0; A01 *= s0; A02 *= s0;
    A10 *= s1; A11 *= s1; A12 *= s1;
    A20 *= s2; A21 *= s2; A22 *= s2;
    const float t0 = c0 / (A00 + A10 + A20 + SK_EPS);
    const float t1 = c1 / (A01 + A11 + A21 + SK_EPS);
    const float t2 = c2 / (A02 + A12 + A22 + SK_EPS);
    A00 *= t0; A10 *= t0; A20 *= t0;
    A01 *= t1; A11 *= t1; A21 *= t1;
    A02 *= t2; A12 *= t2; A22 *= t2;
  }

  if (grow < (long long)B) {
    float4* o = (float4*)(out + grow * 16);   // 64B-aligned -> global_store_b128
    o[0] = make_float4(A00, A01, A02, mm0);
    o[1] = make_float4(A10, A11, A12, mm1);
    o[2] = make_float4(A20, A21, A22, 0.0f);
    o[3] = make_float4(ff0, ff1, 0.0f, 0.0f);
    out[(long long)B * 16 + grow] = V;        // V appended after mus
  }
}

extern "C" void kernel_launch(void* const* d_in, const int* in_sizes, int n_in,
                              void* d_out, int out_size, void* d_ws, size_t ws_size,
                              hipStream_t stream) {
  const float* margins = (const float*)d_in[0];
  const float* W1 = (const float*)d_in[1];  const float* b1 = (const float*)d_in[2];
  const float* W2 = (const float*)d_in[3];  const float* b2 = (const float*)d_in[4];
  const float* W3 = (const float*)d_in[5];  const float* b3 = (const float*)d_in[6];
  const float* W4 = (const float*)d_in[7];  const float* b4 = (const float*)d_in[8];
  float* out = (float*)d_out;
  _Float16* wsB = (_Float16*)d_ws;          // 10 frags * 32 lanes * 32B = 10 KB

  const int B = in_sizes[0] / 6;            // 2,097,152 for the harness
  const int blocks = (B + 255) / 256;       // 256 rows per block

  prep_frags_kernel<<<10, 32, 0, stream>>>(W1, W2, W3, W4, wsB);
  sinkhorn_mlp_kernel<<<blocks, 256, 0, stream>>>(margins, b1, b2, b3, b4,
                                                  wsB, out, B);
}
